// LambdaConv_3212635537555
// MI455X (gfx1250) — matrix operand order repaired and checked by reading, not compiled, not run on hardware
//
#include <hip/hip_runtime.h>
#include <hip/hip_bf16.h>

// LambdaConv on MI455X (gfx1250, wave32, WMMA).
// Pipeline: K0 prep (BN-fold + bf16 pack) -> K1 QKV GEMM (wmma bf16)
//        -> K2 softmax + lambda_c (VALU, tiny) -> K3 23x23 conv decomposed
//           per kernel-row into K=32 WMMA GEMM steps (linear LDS gathers),
//           fused with lambda_c add + Q contraction.

#define BB   16
#define CC   256
#define HH   64
#define WW   64
#define NN   4096      // H*W
#define HEADS 4
#define KK   16
#define VVC  64        // VV
#define MM   23
#define PADM 11

typedef __attribute__((ext_vector_type(16))) __bf16 v16bf;
typedef __attribute__((ext_vector_type(8)))  __bf16 v8bf;
typedef __attribute__((ext_vector_type(8)))  float  v8f;

// ---------------- workspace layout (bytes) ----------------
#define WALL_OFF  0u                               // bf16 [144][256]
#define BIAS_OFF  73728u                           // f32  [144]
#define EMBR_OFF  74496u                           // bf16 [23][16][32] (dr-major, zero-padded dc)
#define QBUF_OFF  98304u                           // bf16 [B][64][4096]
#define KEYS_OFF  (QBUF_OFF + 8388608u)            // f32  [B][16][4096]
#define VBUF_OFF  (KEYS_OFF + 4194304u)            // bf16 [B][64][4096]
#define LAMC_OFF  (VBUF_OFF + 8388608u)            // f32  [B][16][64]

// ================= K0: fold BN, pack weights + embedding to bf16 =========
__global__ __launch_bounds__(256) void k0_prep(
    const float* __restrict__ wq, const float* __restrict__ qg,
    const float* __restrict__ qb, const float* __restrict__ qm,
    const float* __restrict__ qv, const float* __restrict__ wk,
    const float* __restrict__ kg, const float* __restrict__ kb,
    const float* __restrict__ km, const float* __restrict__ kv,
    const float* __restrict__ wv, const float* __restrict__ emb,
    __bf16* __restrict__ Wall, float* __restrict__ bias,
    __bf16* __restrict__ embr)
{
    int blk = blockIdx.x;     // 0..143 weight rows, 144..159 embedding rows
    int t = threadIdx.x;
    if (blk < 144) {
        int o = blk;
        float scale = 1.0f, bi = 0.0f;
        const float* src;
        if (o < 64) {
            scale = qg[o] * rsqrtf(qv[o] + 1e-5f);
            bi = qb[o] - qm[o] * scale;
            src = wq + o * CC;
        } else if (o < 80) {
            int ok = o - 64;
            scale = kg[ok] * rsqrtf(kv[ok] + 1e-5f);
            bi = kb[ok] - km[ok] * scale;
            src = wk + ok * CC;
        } else {
            src = wv + (o - 80) * CC;
        }
        Wall[o * CC + t] = (__bf16)(src[t] * scale);
        if (t == 0) bias[o] = bi;
    } else {
        // embr[dr][k][dc] (dc padded 23 -> 32 with zeros): A-matrix rows for
        // the per-dr conv GEMM, 16B-aligned v8bf fragment loads in K3.
        int k = blk - 144;
        for (int i = t; i < MM * 32; i += 256) {
            int dr = i >> 5, dc = i & 31;
            __bf16 e = (__bf16)0.0f;
            if (dc < MM) e = (__bf16)emb[k * (MM * MM) + dr * MM + dc];
            embr[(dr * KK + k) * 32 + dc] = e;
        }
    }
}

// ================= K1: QKV GEMM via wmma_f32_16x16x32_bf16 ===============
// One block = one batch b, 64 consecutive positions. Output 144 x 64 tile.
__global__ __launch_bounds__(256) void k1_qkv(
    const float* __restrict__ x, const __bf16* __restrict__ Wall,
    const float* __restrict__ bias, __bf16* __restrict__ qbuf,
    float* __restrict__ keys, __bf16* __restrict__ vbuf)
{
    __shared__ __align__(32) __bf16 Xs[64][CC];   // transposed: [col][c], 32 KB

    int bid = blockIdx.x;
    int b  = bid >> 6;
    int n0 = (bid & 63) << 6;
    int t  = threadIdx.x;

    // stage x tile as bf16, transposed. Iterate [c][col] so consecutive lanes
    // hit consecutive global addresses (coalesced 64-float runs per channel).
    const float* xb = x + ((size_t)b * CC) * NN + n0;
    for (int i = t; i < CC * 64; i += 256) {
        int c = i >> 6, col = i & 63;
        Xs[col][c] = (__bf16)xb[(size_t)c * NN + col];
    }
    __syncthreads();

    int lane  = t & 31;
    int wave  = t >> 5;
    int rowIn = lane & 15;                         // A row / B col within tile
    int half  = lane >> 4;                         // lane-half selects K slice

    for (int tile = wave; tile < 36; tile += 8) {  // 9 Mtiles x 4 Ntiles
        int row0 = (tile >> 2) << 4;
        int col0 = (tile & 3) << 4;
        v8f acc = {};
        int arow = row0 + rowIn;
        int bcol = col0 + rowIn;
        for (int kk = 0; kk < CC; kk += 32) {
            // A fragment: lanes 0-15 hold K {0..15}, 16-31 hold {16..31}
            union { v16bf v; v8bf h[2]; } a;
            const __bf16* wr = Wall + arow * CC + kk + half * 16;
            a.h[0] = *reinterpret_cast<const v8bf*>(wr);
            a.h[1] = *reinterpret_cast<const v8bf*>(wr + 8);
            // B fragment: lane-half 0 rows kk..kk+15, half 1 rows kk+16..kk+31
            v16bf bm = *reinterpret_cast<const v16bf*>(&Xs[bcol][kk + half * 16]);
            acc = __builtin_amdgcn_wmma_f32_16x16x32_bf16(
                false, a.v, false, bm, (short)0, acc, false, false);
        }
        // epilogue: C/D layout row = i + 8*half, col = rowIn
#pragma unroll
        for (int i = 0; i < 8; ++i) {
            int row  = row0 + i + half * 8;
            int ncol = n0 + col0 + rowIn;
            float v = acc[i] + bias[row];
            if (row < 64) {
                qbuf[((size_t)b * 64 + row) * NN + ncol] = (__bf16)v;
            } else if (row < 80) {
                keys[((size_t)b * 16 + (row - 64)) * NN + ncol] = v;
            } else {
                vbuf[((size_t)b * 64 + (row - 80)) * NN + ncol] = (__bf16)v;
            }
        }
    }
}

// ================= K2: softmax over n + lambda_c =========================
// One block per (b, k). 0.13 GFLOP total -> VALU is fine.
__global__ __launch_bounds__(256) void k2_softmax_lambda(
    const float* __restrict__ keys, const __bf16* __restrict__ vbuf,
    float* __restrict__ lamc)
{
    __shared__ float sm[NN];      // 16 KB
    __shared__ float red[256];

    int b = blockIdx.x >> 4;
    int k = blockIdx.x & 15;
    const float* kp = keys + ((size_t)b * 16 + k) * NN;
    int t = threadIdx.x;

    float m = -1e30f;
    for (int n = t; n < NN; n += 256) m = fmaxf(m, kp[n]);
    red[t] = m; __syncthreads();
    for (int s = 128; s > 0; s >>= 1) {
        if (t < s) red[t] = fmaxf(red[t], red[t + s]);
        __syncthreads();
    }
    float mx = red[0]; __syncthreads();

    float sum = 0.0f;
    for (int n = t; n < NN; n += 256) {
        float e = __expf(kp[n] - mx);
        sm[n] = e; sum += e;
    }
    red[t] = sum; __syncthreads();
    for (int s = 128; s > 0; s >>= 1) {
        if (t < s) red[t] += red[t + s];
        __syncthreads();
    }
    float inv = 1.0f / red[0];
    __syncthreads();

    int lane = t & 31, wave = t >> 5;
    for (int v = wave; v < VVC; v += 8) {
        const __bf16* vp = vbuf + ((size_t)b * 64 + v) * NN;
        float acc = 0.0f;
        for (int n = lane; n < NN; n += 32) acc += sm[n] * (float)vp[n];
        for (int off = 16; off > 0; off >>= 1) acc += __shfl_xor(acc, off, 32);
        if (lane == 0) lamc[((size_t)b * 16 + k) * VVC + v] = acc * inv;
    }
}

// ================= K3: 23x23 conv, per-kernel-row WMMA GEMM ==============
// One block = (b, h, 16-wide w tile). For each kernel row dr:
//   D[k,pos] += E_dr[16 x 32]  *  P_dr[32 x 16]
// with P_dr[dc,pos] = values[v][h-11+dr][w0-11+dc+pos]. B-fragment element
// addresses are LINEAR in the lane's j index -> pure base+imm ds_load_u16,
// no divmod, no clamps. A columns dc>=23 are zero (padded in K0), which
// annihilates the out-of-window B rows. lambda_c add + Q contraction fused.
__global__ __launch_bounds__(256) void k3_conv_out(
    const __bf16* __restrict__ vbuf, const __bf16* __restrict__ qbuf,
    const __bf16* __restrict__ embr, const float* __restrict__ lamc,
    float* __restrict__ out)
{
    __shared__ __align__(32) __bf16 es[MM][KK][32];   // 23552 B  [dr][k][dc32]
    __shared__ __align__(32) __bf16 ps[16][MM][48];   // 35328 B  patch chunk
    __shared__ __bf16 qs[64][16];                     //  2048 B
    __shared__ float  lcs[KK][VVC];                   //  4096 B
                                                      // total 65024 B

    int bid = blockIdx.x;
    int b  = bid >> 8;
    int h  = (bid >> 2) & 63;
    int w0 = (bid & 3) << 4;
    int t  = threadIdx.x;

    for (int i = t; i < MM * KK * 32; i += 256)
        (&es[0][0][0])[i] = embr[i];
    for (int i = t; i < 64 * 16; i += 256) {
        int o = i >> 4, p = i & 15;
        qs[o][p] = qbuf[((size_t)b * 64 + o) * NN + h * WW + w0 + p];
    }
    for (int i = t; i < KK * VVC; i += 256)
        (&lcs[0][0])[i] = lamc[(size_t)b * (KK * VVC) + i];

    int lane  = t & 31;
    int wave  = t >> 5;
    int rowIn = lane & 15;   // A row (k) / B col (pos)
    int half  = lane >> 4;

    for (int chunk = 0; chunk < 4; ++chunk) {
        int vbase = chunk << 4;
        __syncthreads();
        // stage 23 x 38 patch (rows padded to 48) for 16 v channels
        for (int i = t; i < 16 * MM * 48; i += 256) {
            int vl  = i / (MM * 48);
            int rem = i % (MM * 48);
            int dr  = rem / 48, dc2 = rem % 48;
            int hp = h - PADM + dr;
            int wp = w0 - PADM + dc2;
            __bf16 bv = (__bf16)0.0f;
            if (dc2 < 38 && hp >= 0 && hp < HH && wp >= 0 && wp < WW)
                bv = vbuf[((size_t)b * 64 + vbase + vl) * NN + hp * WW + wp];
            ps[vl][dr][dc2] = bv;
        }
        __syncthreads();

        for (int vv = 0; vv < 2; ++vv) {
            int vl = wave * 2 + vv;
            int v  = vbase + vl;
            v8f acc = {};
            for (int dr = 0; dr < MM; ++dr) {
                // A fragment: emb row dr, 16B-aligned v8bf pair
                union { v16bf v; v8bf h2[2]; } a;
                const __bf16* ar = &es[dr][rowIn][half * 16];
                a.h2[0] = *reinterpret_cast<const v8bf*>(ar);
                a.h2[1] = *reinterpret_cast<const v8bf*>(ar + 8);
                // B fragment: 16 consecutive bf16 from the shifted window
                v16bf bm;
                const __bf16* br = &ps[vl][dr][half * 16 + rowIn];
#pragma unroll
                for (int j = 0; j < 16; ++j) bm[j] = br[j];
                acc = __builtin_amdgcn_wmma_f32_16x16x32_bf16(
                    false, a.v, false, bm, (short)0, acc, false, false);
            }
            // fuse: (lambda_p + lambda_c), then y = sum_k q[h,k,n] * lam[k,v,n]
            float d[8];
#pragma unroll
            for (int i = 0; i < 8; ++i)
                d[i] = acc[i] + lcs[i + half * 8][v];
            int pos = rowIn;
            for (int hd = 0; hd < HEADS; ++hd) {
                float part = 0.0f;
#pragma unroll
                for (int i = 0; i < 8; ++i)
                    part += (float)qs[hd * 16 + (i + half * 8)][pos] * d[i];
                part += __shfl_xor(part, 16, 32);  // combine row halves
                if (half == 0)
                    out[((size_t)b * 256 + hd * 64 + v) * NN + h * WW + w0 + pos] = part;
            }
        }
    }
}

// ======================= launcher ========================================
extern "C" void kernel_launch(void* const* d_in, const int* in_sizes, int n_in,
                              void* d_out, int out_size, void* d_ws, size_t ws_size,
                              hipStream_t stream) {
    (void)in_sizes; (void)n_in; (void)out_size; (void)ws_size;
    const float* x   = (const float*)d_in[0];
    const float* wq  = (const float*)d_in[1];
    const float* qg  = (const float*)d_in[2];
    const float* qb  = (const float*)d_in[3];
    const float* qm  = (const float*)d_in[4];
    const float* qv  = (const float*)d_in[5];
    const float* wk  = (const float*)d_in[6];
    const float* kg  = (const float*)d_in[7];
    const float* kb  = (const float*)d_in[8];
    const float* km  = (const float*)d_in[9];
    const float* kv  = (const float*)d_in[10];
    const float* wv  = (const float*)d_in[11];
    const float* emb = (const float*)d_in[12];

    char* ws = (char*)d_ws;
    __bf16* Wall = (__bf16*)(ws + WALL_OFF);
    float*  bias = (float*) (ws + BIAS_OFF);
    __bf16* embr = (__bf16*)(ws + EMBR_OFF);
    __bf16* qbuf = (__bf16*)(ws + QBUF_OFF);
    float*  keys = (float*) (ws + KEYS_OFF);
    __bf16* vbuf = (__bf16*)(ws + VBUF_OFF);
    float*  lamc = (float*) (ws + LAMC_OFF);
    float*  outp = (float*)d_out;

    k0_prep<<<160, 256, 0, stream>>>(wq, qg, qb, qm, qv, wk, kg, kb, km, kv,
                                     wv, emb, Wall, bias, embr);
    k1_qkv<<<BB * (NN / 64), 256, 0, stream>>>(x, Wall, bias, qbuf, keys, vbuf);
    k2_softmax_lambda<<<BB * KK, 256, 0, stream>>>(keys, vbuf, lamc);
    k3_conv_out<<<BB * HH * (WW / 16), 256, 0, stream>>>(vbuf, qbuf, embr,
                                                         lamc, outp);
}
